// DM_no_aux_44504451121739
// MI455X (gfx1250) — matrix operand order repaired and checked by reading, not compiled, not run on hardware
//
#include <hip/hip_runtime.h>
#include <stdint.h>

// ---------------- problem constants ----------------
constexpr int NB   = 8;      // batch
constexpr int NS   = 2048;   // seq
constexpr int ND   = 256;    // model dim
constexpr int NH   = 8;      // heads
constexpr int NHD  = 32;     // head dim
constexpr int NMLP = 1024;   // mlp dim
constexpr int NKP  = 512;    // k-predictor hidden
constexpr int BSs  = NB * NS;
constexpr size_t SD = (size_t)NS * ND;

typedef __bf16 bf16;
typedef __attribute__((ext_vector_type(16))) __bf16 v16bf;
typedef __attribute__((ext_vector_type(8)))  float  v8f;
typedef __attribute__((ext_vector_type(4)))  unsigned int u32x4;
typedef __attribute__((ext_vector_type(8)))  int    i32x8;
typedef __attribute__((ext_vector_type(4)))  int    i32x4;

#if defined(__HIP_DEVICE_COMPILE__) && __has_builtin(__builtin_amdgcn_tensor_load_to_lds) && __has_builtin(__builtin_amdgcn_s_wait_tensorcnt)
#define USE_TDM 1
#else
#define USE_TDM 0
#endif

// ---------------- wave helpers (wave32) ----------------
__device__ inline float wred_sum(float v) {
  #pragma unroll
  for (int m = 16; m >= 1; m >>= 1) v += __shfl_xor(v, m, 32);
  return v;
}
__device__ inline float hred_max(float v) {  // reduce within 16-lane half
  #pragma unroll
  for (int m = 8; m >= 1; m >>= 1) v = fmaxf(v, __shfl_xor(v, m, 32));
  return v;
}
// A-fragment (16x32 bf16) K index pattern for lane-half hi, vgpr-pair p
__device__ inline int afrag_k(int p, int hi) {
  return hi * 16 + 2 * p;
}
__device__ inline float gelu_tanh(float x) {
  float x3 = x * x * x;
  float t = tanhf(0.7978845608028654f * (x + 0.044715f * x3));
  return 0.5f * x * (1.f + t);
}

// ---------------- router: hidden init ----------------
__global__ void k_init_hidden(float* __restrict__ hidden, const float* __restrict__ kp_b1) {
  int i = blockIdx.x * 256 + threadIdx.x;  // 4096
  hidden[i] = kp_b1[i & (NKP - 1)];
}

// ---------------- router: split-K kp_w1 GEMV (bandwidth kernel) ----------------
__global__ void __launch_bounds__(256) k_kp_splitk(const float* __restrict__ x,
                                                   const float* __restrict__ w1,
                                                   float* __restrict__ hidden) {
  __shared__ float xl[8][128];
  const int tid = threadIdx.x;
  const size_t i0 = (size_t)blockIdx.x * 2048;  // 256 blocks x 2048 rows
  float acc0[8], acc1[8];
  #pragma unroll
  for (int b = 0; b < 8; b++) { acc0[b] = 0.f; acc1[b] = 0.f; }
  for (int c0 = 0; c0 < 2048; c0 += 128) {
    __syncthreads();
    #pragma unroll
    for (int t = 0; t < 4; t++) {
      int idx = tid + t * 256; int b = idx >> 7; int i = idx & 127;
      xl[b][i] = x[(size_t)b * SD + i0 + c0 + i];
    }
    __syncthreads();
    for (int i = 0; i < 128; i++) {
      size_t row = (i0 + c0 + i) * NKP;
      float w0 = w1[row + tid];
      float w1v = w1[row + 256 + tid];
      #pragma unroll
      for (int b = 0; b < 8; b++) {
        float xv = xl[b][i];
        acc0[b] = fmaf(xv, w0, acc0[b]);
        acc1[b] = fmaf(xv, w1v, acc1[b]);
      }
    }
  }
  #pragma unroll
  for (int b = 0; b < 8; b++) {
    atomicAdd(&hidden[b * NKP + tid], acc0[b]);
    atomicAdd(&hidden[b * NKP + 256 + tid], acc1[b]);
  }
}

// ---------------- router: finalize k per batch ----------------
__global__ void k_finalize(const float* __restrict__ hidden, const float* __restrict__ kp_w2,
                           const float* __restrict__ kp_b2, float* __restrict__ kf) {
  __shared__ float red[16];
  int tid = threadIdx.x;  // 512
  for (int b = 0; b < 8; b++) {
    float h = hidden[b * NKP + tid];
    float lr = h > 0.f ? h : 0.01f * h;         // leaky_relu 0.01
    float c = lr * kp_w2[tid];
    float ws = wred_sum(c);
    if ((tid & 31) == 0) red[tid >> 5] = ws;
    __syncthreads();
    if (tid < 16) {
      float v = red[tid];
      #pragma unroll
      for (int m = 8; m >= 1; m >>= 1) v += __shfl_xor(v, m, 32);
      if (tid == 0) {
        float kl = v + kp_b2[0];
        float sig = 1.f / (1.f + __expf(-kl));
        float kv = fminf(fmaxf(sig * (float)NS, 1.f), (float)NS);
        kf[b] = (float)((int)kv);               // floor (.astype int32)
      }
    }
    __syncthreads();
  }
}

// ---------------- LayerNorm (wave per row) + optional router weights ----------------
template <bool ROUTER>
__global__ void __launch_bounds__(256) k_ln(const float* __restrict__ in,
                                            const float* __restrict__ g, const float* __restrict__ bta,
                                            bf16* __restrict__ outb,
                                            const float* __restrict__ wp_w, const float* __restrict__ wp_b,
                                            float* __restrict__ weights) {
  int wid = threadIdx.x >> 5, lane = threadIdx.x & 31;
  int row = blockIdx.x * 8 + wid;
  const float* xr = in + (size_t)row * ND;
  float v[8];
  float s = 0.f, s2 = 0.f, dot = 0.f;
  #pragma unroll
  for (int t = 0; t < 8; t++) {
    v[t] = xr[lane + t * 32];
    s += v[t]; s2 += v[t] * v[t];
    if (ROUTER) dot += v[t] * wp_w[lane + t * 32];
  }
  s = wred_sum(s); s2 = wred_sum(s2);
  float mean = s * (1.f / ND);
  float var = s2 * (1.f / ND) - mean * mean;
  float rstd = rsqrtf(var + 1e-5f);
  #pragma unroll
  for (int t = 0; t < 8; t++) {
    int d = lane + t * 32;
    outb[(size_t)row * ND + d] = (bf16)((v[t] - mean) * rstd * g[d] + bta[d]);
  }
  if (ROUTER) {
    dot = wred_sum(dot);
    if (lane == 0) weights[row] = dot + wp_b[0];
  }
}

// ---------------- sel / additive key mask ----------------
__global__ void k_masks(const float* __restrict__ weights, const float* __restrict__ kf,
                        const float* __restrict__ amask, float* __restrict__ sel,
                        float* __restrict__ keymask) {
  int idx = blockIdx.x * 256 + threadIdx.x;  // BSs
  int b = idx >> 11;
  float w = weights[idx];
  float sl = (w > kf[b]) ? 1.f : 0.f;
  sel[idx] = sl;
  keymask[idx] = amask[idx] + (sl > 0.f ? 0.f : -1e9f);
}

// ---------------- f32 -> bf16 cast ----------------
__global__ void k_cast(const float* __restrict__ in, bf16* __restrict__ out, int n) {
  int i = blockIdx.x * 256 + threadIdx.x;
  if (i < n) out[i] = (bf16)in[i];
}

// ---------------- generic bf16 WMMA GEMM, 64x64 block tile, 8 waves ----------------
#define GM_QKV 0
#define GM_RES 1
#define GM_GELU 2
#define GM_FINAL 3

template <int MODE>
__global__ void __launch_bounds__(256) k_gemm(
    const bf16* __restrict__ A, const bf16* __restrict__ W, const float* __restrict__ bias,
    int M, int N, int K,
    bf16* __restrict__ qo, bf16* __restrict__ ko, bf16* __restrict__ vo,   // QKV scatter
    const float* __restrict__ xres, float* __restrict__ aout,              // RES
    bf16* __restrict__ mbo,                                                // GELU
    const float* __restrict__ apre, const float* __restrict__ wgt,
    const float* __restrict__ sel, float* __restrict__ fout)               // FINAL
{
  __shared__ bf16 At[64][34];   // padded: conflict-free A-frag gathers
  __shared__ bf16 Bt[32][64];
  const int tid = threadIdx.x;
  const int lane = tid & 31, wid = tid >> 5;
  const int hi = lane >> 4, ln = lane & 15;
  const int nblk = N >> 6;
  const int bm = (blockIdx.x / nblk) * 64;
  const int bn = (blockIdx.x % nblk) * 64;
  const int msub = (wid & 3) * 16;
  const int nsub = (wid >> 2) * 32;

  v8f acc0 = {}; v8f acc1 = {};
  for (int k0 = 0; k0 < K; k0 += 32) {
    __syncthreads();
    #pragma unroll
    for (int t = 0; t < 2; t++) {                     // A tile 64x32
      int idx = tid + t * 256;
      int r = idx >> 3; int c = (idx & 7) * 4;
      const bf16* src = A + (size_t)(bm + r) * K + k0 + c;
      #pragma unroll
      for (int u = 0; u < 4; u++) At[r][c + u] = src[u];
    }
    #pragma unroll
    for (int t = 0; t < 2; t++) {                     // B tile 32x64
      int idx = tid + t * 256;
      int r = idx >> 4; int c = (idx & 15) * 4;
      const bf16* src = W + (size_t)(k0 + r) * N + bn + c;
      #pragma unroll
      for (int u = 0; u < 4; u++) Bt[r][c + u] = src[u];
    }
    __syncthreads();
    v16bf a, b0, b1;
    #pragma unroll
    for (int p = 0; p < 8; p++) {
      int kk = afrag_k(p, hi);
      a[2 * p]     = At[msub + ln][kk];
      a[2 * p + 1] = At[msub + ln][kk + 1];
    }
    #pragma unroll
    for (int h2 = 0; h2 < 16; h2++) {
      int kk = hi * 16 + h2;
      b0[h2] = Bt[kk][nsub + ln];
      b1[h2] = Bt[kk][nsub + 16 + ln];
    }
    acc0 = __builtin_amdgcn_wmma_f32_16x16x32_bf16(false, a, false, b0, (short)0, acc0, false, false);
    acc1 = __builtin_amdgcn_wmma_f32_16x16x32_bf16(false, a, false, b1, (short)0, acc1, false, false);
  }
  // epilogue: lane holds C[bm+msub+hi*8+r][bn+nsub+(half*16)+ln]
  #pragma unroll
  for (int r = 0; r < 8; r++) {
    int row = bm + msub + hi * 8 + r;
    #pragma unroll
    for (int half = 0; half < 2; half++) {
      int col = bn + nsub + half * 16 + ln;
      float v = (half ? acc1[r] : acc0[r]) + bias[col];
      if (MODE == GM_QKV) {
        int which = col >> 8, rem = col & 255;
        int h = rem >> 5, hd = rem & 31;
        int b = row >> 11, s = row & 2047;
        size_t dst = ((size_t)(b * NH + h) * NS + s) * NHD + hd;
        bf16* p = (which == 0) ? qo : (which == 1) ? ko : vo;
        p[dst] = (bf16)v;
      } else if (MODE == GM_RES) {
        size_t off = (size_t)row * ND + col;
        aout[off] = xres[off] + v;
      } else if (MODE == GM_GELU) {
        mbo[(size_t)row * NMLP + col] = (bf16)gelu_tanh(v);
      } else {  // GM_FINAL: out = x + sel * weight * (a + mlp)
        size_t off = (size_t)row * ND + col;
        float blk = apre[off] + v;
        fout[off] = xres[off] + sel[row] * wgt[row] * blk;
      }
    }
  }
}

// ---------------- TDM: 1D contiguous global -> LDS with pad every 64B ----------------
#if USE_TDM
__device__ inline void tdm_load_padded(unsigned lds_off, const void* gsrc, unsigned nelems2B) {
  unsigned long long ga = (unsigned long long)(uintptr_t)gsrc;
  u32x4 g0;
  g0[0] = 1u;                                   // count=1 valid descriptor
  g0[1] = lds_off;                              // lds_addr (bytes)
  g0[2] = (unsigned)(ga & 0xffffffffu);         // global_addr lo
  g0[3] = (unsigned)((ga >> 32) & 0x01ffffffu) | (2u << 30);  // addr hi | type=2
  i32x8 g1;
  // data_size=1 (2B), pad_enable, pad_interval=3 (16 dwords = 64B), pad_amount=1 (2 dwords = 8B)
  g1[0] = (int)((1u << 16) | (1u << 20) | (3u << 22) | (1u << 25));
  g1[1] = (int)((nelems2B & 0xffffu) << 16);            // tensor_dim0[15:0]
  g1[2] = (int)(((nelems2B >> 16) & 0xffffu) | (1u << 16)); // tensor_dim0 hi | tensor_dim1=1
  g1[3] = (int)((nelems2B & 0xffffu) << 16);            // tile_dim0
  g1[4] = 0;                                            // tile_dim1=0 (1D), tile_dim2=0
  g1[5] = (int)nelems2B;                                // tensor_dim0_stride lo
  g1[6] = 0; g1[7] = 0;
  i32x4 z4 = {0, 0, 0, 0};
  i32x8 z8 = {0, 0, 0, 0, 0, 0, 0, 0};
  // clang-23 / therock form: (g0, g1, g2, g3, g4, cpol)
  __builtin_amdgcn_tensor_load_to_lds(g0, g1, z4, z4, z8, 0);
}
#endif

// ---------------- flash attention: 64-key blocks, WMMA row-sums, TDM-staged K/V ----------------
__global__ void __launch_bounds__(256) k_attn(const bf16* __restrict__ qb,
                                              const bf16* __restrict__ kbuf,
                                              const bf16* __restrict__ vbuf,
                                              const float* __restrict__ keymask,
                                              bf16* __restrict__ attnb) {
  __shared__ bf16 Kt[64][36];       // padded (72B row) -> conflict-free gathers
  __shared__ bf16 Vt[64][36];
  __shared__ bf16 Pt[8][16][72];    // per-wave P scratch, 64 keys + 8 pad
  const int tid = threadIdx.x;
  const int lane = tid & 31, wid = tid >> 5;
  const int hi = lane >> 4, ln = lane & 15;
  const int nqb = NS / 128;                 // 16
  const int bh = blockIdx.x / nqb;          // 0..63
  const int b = bh / NH, h = bh % NH;
  const int q0 = (blockIdx.x % nqb) * 128 + wid * 16;
  const size_t base = (size_t)bh * NS * NHD;

  // Q fragment for this wave's 16 rows (A 16x32, K-dim == head dim)
  v16bf qa;
  {
    const bf16* qrow = qb + base + (size_t)(q0 + ln) * NHD;
    #pragma unroll
    for (int p = 0; p < 8; p++) {
      int kk = afrag_k(p, hi);
      qa[2 * p] = qrow[kk]; qa[2 * p + 1] = qrow[kk + 1];
    }
  }
  v16bf onesf;
  #pragma unroll
  for (int h2 = 0; h2 < 16; h2++) onesf[h2] = (bf16)1.0f;

  float mrow[8], lrow[8], scr[8];
  #pragma unroll
  for (int r = 0; r < 8; r++) { mrow[r] = -3.0e38f; lrow[r] = 0.f; }
  v8f o0 = {}; v8f o1 = {};
  const float sc = 0.17677669529663687f;    // 1/sqrt(32)

  for (int kb0 = 0; kb0 < NS; kb0 += 64) {
    __syncthreads();                        // previous tile fully consumed
#if USE_TDM
    if (wid == 0) {
      tdm_load_padded((unsigned)(uintptr_t)(void*)&Kt[0][0], kbuf + base + (size_t)kb0 * NHD, 64 * NHD);
      tdm_load_padded((unsigned)(uintptr_t)(void*)&Vt[0][0], vbuf + base + (size_t)kb0 * NHD, 64 * NHD);
      __builtin_amdgcn_s_wait_tensorcnt(0);
    }
#else
    #pragma unroll
    for (int t = 0; t < 8; t++) {
      int idx = tid + t * 256;              // 2048 elems each
      int r = idx >> 5, c = idx & 31;
      Kt[r][c] = kbuf[base + (size_t)(kb0 + r) * NHD + c];
      Vt[r][c] = vbuf[base + (size_t)(kb0 + r) * NHD + c];
    }
#endif
    __syncthreads();

    // ---- scores for 64 keys: 4 WMMAs (B-frag = K^T, K-dim = head dim) ----
    v8f s0 = {}, s1 = {}, s2 = {}, s3 = {};
    {
      v16bf kf;
      #pragma unroll
      for (int h2 = 0; h2 < 16; h2++) kf[h2] = Kt[ln][hi * 16 + h2];
      s0 = __builtin_amdgcn_wmma_f32_16x16x32_bf16(false, qa, false, kf, (short)0, s0, false, false);
      #pragma unroll
      for (int h2 = 0; h2 < 16; h2++) kf[h2] = Kt[16 + ln][hi * 16 + h2];
      s1 = __builtin_amdgcn_wmma_f32_16x16x32_bf16(false, qa, false, kf, (short)0, s1, false, false);
      #pragma unroll
      for (int h2 = 0; h2 < 16; h2++) kf[h2] = Kt[32 + ln][hi * 16 + h2];
      s2 = __builtin_amdgcn_wmma_f32_16x16x32_bf16(false, qa, false, kf, (short)0, s2, false, false);
      #pragma unroll
      for (int h2 = 0; h2 < 16; h2++) kf[h2] = Kt[48 + ln][hi * 16 + h2];
      s3 = __builtin_amdgcn_wmma_f32_16x16x32_bf16(false, qa, false, kf, (short)0, s3, false, false);
    }
    float km0 = keymask[(size_t)b * NS + kb0 + ln];
    float km1 = keymask[(size_t)b * NS + kb0 + 16 + ln];
    float km2 = keymask[(size_t)b * NS + kb0 + 32 + ln];
    float km3 = keymask[(size_t)b * NS + kb0 + 48 + ln];

    // ---- online softmax over 64 keys (row = hi*8+r owned by a 16-lane half) ----
    #pragma unroll
    for (int r = 0; r < 8; r++) {
      float a0 = s0[r] * sc + km0;
      float a1 = s1[r] * sc + km1;
      float a2 = s2[r] * sc + km2;
      float a3 = s3[r] * sc + km3;
      float mx = hred_max(fmaxf(fmaxf(a0, a1), fmaxf(a2, a3)));
      float mnew = fmaxf(mrow[r], mx);
      float scale = __expf(mrow[r] - mnew);
      mrow[r] = mnew; scr[r] = scale;
      o0[r] = o0[r] * scale; o1[r] = o1[r] * scale;
      float p0 = __expf(a0 - mnew), p1 = __expf(a1 - mnew);
      float p2 = __expf(a2 - mnew), p3 = __expf(a3 - mnew);
      int m = hi * 8 + r;
      Pt[wid][m][ln]      = (bf16)p0;
      Pt[wid][m][16 + ln] = (bf16)p1;
      Pt[wid][m][32 + ln] = (bf16)p2;
      Pt[wid][m][48 + ln] = (bf16)p3;
    }
    asm volatile("s_wait_dscnt 0" ::: "memory");  // wave-private LDS RAW

    // ---- P A-frags for keys 0..31 and 32..63 ----
    v16bf pa0, pa1;
    #pragma unroll
    for (int p = 0; p < 8; p++) {
      int kk = afrag_k(p, hi);
      pa0[2 * p] = Pt[wid][ln][kk];       pa0[2 * p + 1] = Pt[wid][ln][kk + 1];
      pa1[2 * p] = Pt[wid][ln][32 + kk];  pa1[2 * p + 1] = Pt[wid][ln][32 + kk + 1];
    }
    // ---- row sums via WMMA with ones B (every column = row sum) ----
    v8f ssum = {};
    ssum = __builtin_amdgcn_wmma_f32_16x16x32_bf16(false, pa0, false, onesf, (short)0, ssum, false, false);
    ssum = __builtin_amdgcn_wmma_f32_16x16x32_bf16(false, pa1, false, onesf, (short)0, ssum, false, false);
    #pragma unroll
    for (int r = 0; r < 8; r++) lrow[r] = lrow[r] * scr[r] + ssum[r];

    // ---- PV: o[16x32] += P[16x64] @ V[64x32], 4 WMMAs ----
    {
      v16bf vf;
      #pragma unroll
      for (int h2 = 0; h2 < 16; h2++) vf[h2] = Vt[hi * 16 + h2][ln];
      o0 = __builtin_amdgcn_wmma_f32_16x16x32_bf16(false, pa0, false, vf, (short)0, o0, false, false);
      #pragma unroll
      for (int h2 = 0; h2 < 16; h2++) vf[h2] = Vt[32 + hi * 16 + h2][ln];
      o0 = __builtin_amdgcn_wmma_f32_16x16x32_bf16(false, pa1, false, vf, (short)0, o0, false, false);
      #pragma unroll
      for (int h2 = 0; h2 < 16; h2++) vf[h2] = Vt[hi * 16 + h2][16 + ln];
      o1 = __builtin_amdgcn_wmma_f32_16x16x32_bf16(false, pa0, false, vf, (short)0, o1, false, false);
      #pragma unroll
      for (int h2 = 0; h2 < 16; h2++) vf[h2] = Vt[32 + hi * 16 + h2][16 + ln];
      o1 = __builtin_amdgcn_wmma_f32_16x16x32_bf16(false, pa1, false, vf, (short)0, o1, false, false);
    }
  }
  // normalize + write [b, s, h*32+d] as bf16 for the output projection GEMM
  #pragma unroll
  for (int r = 0; r < 8; r++) {
    int qrow = q0 + hi * 8 + r;
    float inv = 1.f / lrow[r];
    size_t off = ((size_t)b * NS + qrow) * ND + h * NHD;
    attnb[off + ln] = (bf16)(o0[r] * inv);
    attnb[off + 16 + ln] = (bf16)(o1[r] * inv);
  }
}

// ---------------- host launch ----------------
extern "C" void kernel_launch(void* const* d_in, const int* in_sizes, int n_in,
                              void* d_out, int out_size, void* d_ws, size_t ws_size,
                              hipStream_t stream) {
  (void)in_sizes; (void)n_in; (void)out_size; (void)ws_size;
  const float* x      = (const float*)d_in[0];
  const float* amask  = (const float*)d_in[1];
  const float* wp_w   = (const float*)d_in[2];
  const float* wp_b   = (const float*)d_in[3];
  const float* kp_w1  = (const float*)d_in[4];
  const float* kp_b1  = (const float*)d_in[5];
  const float* kp_w2  = (const float*)d_in[6];
  const float* kp_b2  = (const float*)d_in[7];
  const float* ln1_g  = (const float*)d_in[8];
  const float* ln1_b  = (const float*)d_in[9];
  const float* wqkv   = (const float*)d_in[10];
  const float* bqkv   = (const float*)d_in[11];
  const float* wo     = (const float*)d_in[12];
  const float* bo     = (const float*)d_in[13];
  const float* ln2_g  = (const float*)d_in[14];
  const float* ln2_b  = (const float*)d_in[15];
  const float* wm1    = (const float*)d_in[16];
  const float* bm1    = (const float*)d_in[17];
  const float* wm2    = (const float*)d_in[18];
  const float* bm2    = (const float*)d_in[19];
  float* out = (float*)d_out;

  char* ws = (char*)d_ws;
  size_t off = 0;
  auto alloc = [&](size_t bytes) { size_t o = off; off += (bytes + 255) & ~(size_t)255; return o; };
  float* hidden  = (float*)(ws + alloc((size_t)NB * NKP * 4));
  float* kf      = (float*)(ws + alloc(NB * 4));
  float* weights = (float*)(ws + alloc((size_t)BSs * 4));
  float* sel     = (float*)(ws + alloc((size_t)BSs * 4));
  float* keymask = (float*)(ws + alloc((size_t)BSs * 4));
  bf16* wqkvb    = (bf16*)(ws + alloc((size_t)ND * 3 * ND * 2));
  bf16* wob      = (bf16*)(ws + alloc((size_t)ND * ND * 2));
  bf16* wm1b     = (bf16*)(ws + alloc((size_t)ND * NMLP * 2));
  bf16* wm2b     = (bf16*)(ws + alloc((size_t)NMLP * ND * 2));
  size_t actb = (size_t)BSs * ND * 2;           // 8 MB each, 256-aligned
  bf16* h1b   = (bf16*)(ws + alloc(actb));
  bf16* qbuf  = (bf16*)(ws + alloc(actb));
  bf16* kbuf  = (bf16*)(ws + alloc(actb));
  bf16* vbuf  = (bf16*)(ws + alloc(actb));
  bf16* attnb = (bf16*)(ws + alloc(actb));
  float* aout = (float*)(ws + alloc((size_t)BSs * ND * 4));
  bf16* h2b   = (bf16*)(ws + alloc(actb));
  bf16* mb    = h1b;                            // alias: h1/q/k/v dead by MLP1 (32 MB)

  // ---- token router ----
  k_init_hidden<<<(NB * NKP) / 256, 256, 0, stream>>>(hidden, kp_b1);
  k_kp_splitk<<<(int)(SD / 2048), 256, 0, stream>>>(x, kp_w1, hidden);
  k_finalize<<<1, 512, 0, stream>>>(hidden, kp_w2, kp_b2, kf);
  // ---- LN1 + per-token router weights ----
  k_ln<true><<<BSs / 8, 256, 0, stream>>>(x, ln1_g, ln1_b, h1b, wp_w, wp_b, weights);
  k_masks<<<BSs / 256, 256, 0, stream>>>(weights, kf, amask, sel, keymask);
  // ---- cast weight matrices to bf16 ----
  k_cast<<<(ND * 3 * ND) / 256, 256, 0, stream>>>(wqkv, wqkvb, ND * 3 * ND);
  k_cast<<<(ND * ND) / 256, 256, 0, stream>>>(wo, wob, ND * ND);
  k_cast<<<(ND * NMLP) / 256, 256, 0, stream>>>(wm1, wm1b, ND * NMLP);
  k_cast<<<(NMLP * ND) / 256, 256, 0, stream>>>(wm2, wm2b, NMLP * ND);
  // ---- QKV projection ----
  k_gemm<GM_QKV><<<(BSs / 64) * (3 * ND / 64), 256, 0, stream>>>(
      h1b, wqkvb, bqkv, BSs, 3 * ND, ND,
      qbuf, kbuf, vbuf, nullptr, nullptr, nullptr, nullptr, nullptr, nullptr, nullptr);
  // ---- masked flash attention ----
  k_attn<<<NB * NH * (NS / 128), 256, 0, stream>>>(qbuf, kbuf, vbuf, keymask, attnb);
  // ---- output projection + residual ----
  k_gemm<GM_RES><<<(BSs / 64) * (ND / 64), 256, 0, stream>>>(
      attnb, wob, bo, BSs, ND, ND,
      nullptr, nullptr, nullptr, x, aout, nullptr, nullptr, nullptr, nullptr, nullptr);
  // ---- LN2 ----
  k_ln<false><<<BSs / 8, 256, 0, stream>>>(aout, ln2_g, ln2_b, h2b, nullptr, nullptr, nullptr);
  // ---- MLP1 + gelu ----
  k_gemm<GM_GELU><<<(BSs / 64) * (NMLP / 64), 256, 0, stream>>>(
      h2b, wm1b, bm1, BSs, NMLP, ND,
      nullptr, nullptr, nullptr, nullptr, nullptr, mb, nullptr, nullptr, nullptr, nullptr);
  // ---- MLP2 + residual + routed combine: out = x + sel*w*(a + mlp) ----
  k_gemm<GM_FINAL><<<(BSs / 64) * (ND / 64), 256, 0, stream>>>(
      mb, wm2b, bm2, BSs, ND, NMLP,
      nullptr, nullptr, nullptr, x, nullptr, nullptr, aout, weights, sel, out);
}
